// BDCM_56538949484820
// MI455X (gfx1250) — matrix operand order repaired and checked by reading, not compiled, not run on hardware
//
#include <hip/hip_runtime.h>
#include <hip/hip_bf16.h>

typedef __attribute__((ext_vector_type(16))) __bf16 v16bf;
typedef __attribute__((ext_vector_type(8)))  float  v8f;
typedef __attribute__((ext_vector_type(4)))  float  f4;

// B=8, C=256, H=W=64, HW=4096, NTAP=9, K=2304, CO=256

// ---------------- per-(b,c) mean/max reduction ----------------
__global__ void BDCM_stats_kernel(const float* __restrict__ x,
                                  float* __restrict__ avg, float* __restrict__ mx) {
    __shared__ float ssum[256], smax[256];
    int bc = blockIdx.x;
    const float* p = x + (size_t)bc * 4096;
    float s = 0.f, m = -3.4e38f;
    for (int i = threadIdx.x; i < 4096; i += 256) { float v = p[i]; s += v; m = fmaxf(m, v); }
    ssum[threadIdx.x] = s; smax[threadIdx.x] = m;
    __syncthreads();
    for (int off = 128; off > 0; off >>= 1) {
        if (threadIdx.x < off) {
            ssum[threadIdx.x] += ssum[threadIdx.x + off];
            smax[threadIdx.x] = fmaxf(smax[threadIdx.x], smax[threadIdx.x + off]);
        }
        __syncthreads();
    }
    if (threadIdx.x == 0) { avg[bc] = ssum[0] * (1.f / 4096.f); mx[bc] = smax[0]; }
}

// ---------------- channel-gate MLP + sigmoid ----------------
__global__ void BDCM_gate_kernel(const float* __restrict__ avg, const float* __restrict__ mx,
                                 const float* __restrict__ w1, const float* __restrict__ b1,
                                 const float* __restrict__ w2, const float* __restrict__ b2,
                                 float* __restrict__ att) {
    __shared__ float sa[256], sm[256], hsum[16];
    int b = blockIdx.x;
    sa[threadIdx.x] = avg[b * 256 + threadIdx.x];
    sm[threadIdx.x] = mx[b * 256 + threadIdx.x];
    __syncthreads();
    if (threadIdx.x < 16) {
        int j = threadIdx.x;
        float ha = b1[j], hm = b1[j];
        for (int c = 0; c < 256; ++c) { float w = w1[j * 256 + c]; ha += w * sa[c]; hm += w * sm[c]; }
        hsum[j] = fmaxf(ha, 0.f) + fmaxf(hm, 0.f);
    }
    __syncthreads();
    int c = threadIdx.x;
    float s = 2.f * b2[c];
    #pragma unroll
    for (int j = 0; j < 16; ++j) s += w2[c * 16 + j] * hsum[j];
    att[b * 256 + c] = 1.f / (1.f + __expf(-s));
}

// ---------------- h = x * att (float4) ----------------
__global__ void BDCM_scale_kernel(const float* __restrict__ x, const float* __restrict__ att,
                                  float* __restrict__ h) {
    size_t i4 = (size_t)blockIdx.x * 256 + threadIdx.x;   // 2M float4
    int bc = (int)(i4 >> 10);                              // 1024 float4 per plane
    float a = att[bc];
    f4 v = ((const f4*)x)[i4];
    ((f4*)h)[i4] = v * a;
}

// ---------------- pack dconv weights into WMMA B-fragment layout (bf16) ----
// B frag (K=32 x N=16): lane l -> n = l&15, k-half group = l>>4 ; element e -> k = (l>>4)*16 + e
__global__ void BDCM_packw_kernel(const float* __restrict__ w, __bf16* __restrict__ pw) {
    int id = blockIdx.x * 256 + threadIdx.x;     // 16*72*32*16 = 589824
    int e    = id & 15;
    int lane = (id >> 4) & 31;
    int kt   = (id >> 9) % 72;
    int nt   = id / 36864;
    int o = nt * 16 + (lane & 15);
    int k = kt * 32 + ((lane >> 4) << 4) + e;
    int c = k / 9, tap = k % 9;
    pw[id] = (__bf16)w[(o * 256 + c) * 9 + tap];
}

// ---------------- 3x3 conv 256 -> 18 (offsets), LDS tiled ----------------
__global__ void BDCM_offconv_kernel(const float* __restrict__ h, const float* __restrict__ pw,
                                    const float* __restrict__ pb, float* __restrict__ offs) {
    __shared__ float srow[3][32][66];
    __shared__ float swt[18][32][9];
    int by = blockIdx.x;              // b*64 + y
    int b = by >> 6, y = by & 63;
    int tx = threadIdx.x & 63;
    int q  = threadIdx.x >> 6;        // 0..3
    float acc[5];
    #pragma unroll
    for (int j = 0; j < 5; ++j) { int o = q + 4 * j; acc[j] = (o < 18) ? pb[o] : 0.f; }
    for (int cb = 0; cb < 8; ++cb) {
        __syncthreads();
        for (int i = threadIdx.x; i < 5184; i += 256) {          // 18*32*9 weights
            int o = i / 288, r = i % 288, c = r / 9, t = r % 9;
            swt[o][c][t] = pw[(o * 256 + cb * 32 + c) * 9 + t];
        }
        for (int i = threadIdx.x; i < 6336; i += 256) {          // 3*32*66 pixels
            int ky = i / 2112, r = i % 2112, c = r / 66, xe = r % 66;
            int yy = y + ky - 1, xx = xe - 1;
            float v = 0.f;
            if (yy >= 0 && yy < 64 && xx >= 0 && xx < 64)
                v = h[(((b * 256 + cb * 32 + c) << 6) + yy) * 64 + xx];
            srow[ky][c][xe] = v;
        }
        __syncthreads();
        for (int c = 0; c < 32; ++c)
            #pragma unroll
            for (int ky = 0; ky < 3; ++ky)
                #pragma unroll
                for (int kx = 0; kx < 3; ++kx) {
                    float v = srow[ky][c][tx + kx];
                    #pragma unroll
                    for (int j = 0; j < 5; ++j) {
                        int o = q + 4 * j;
                        if (o < 18) acc[j] += v * swt[o][c][ky * 3 + kx];
                    }
                }
    }
    #pragma unroll
    for (int j = 0; j < 5; ++j) {
        int o = q + 4 * j;
        if (o < 18) offs[(((b * 18 + o) << 6) + y) * 64 + tx] = acc[j];
    }
}

// ---------------- fused bilinear sampling + bf16 WMMA GEMM ----------------
// block = 32 pixels (M) x 256 output channels (N); 8 waves in 2(M) x 4(N)
__global__ void __launch_bounds__(256, 2)
BDCM_deform_wmma_kernel(const float* __restrict__ h, const float* __restrict__ offs,
                        const __bf16* __restrict__ pw, float* __restrict__ out) {
    __shared__ int    sIdx[288 * 4];
    __shared__ float  sWt[288 * 4];
    __shared__ __bf16 sA[2 * 9 * 32 * 16];     // A fragments: [mt][kt][lane][elem]
    const int tid = threadIdx.x;
    const int m0  = blockIdx.x * 32;

    // Per-block sampling metadata: 32 pixels x 9 taps -> 4 corner indices + 4 weights
    for (int i = tid; i < 288; i += 256) {
        int mi = i / 9, n = i % 9;
        int m = m0 + mi;
        int b = m >> 12, pos = m & 4095;
        int y = pos >> 6, x = pos & 63;
        float offx = offs[((b * 18 + n) << 12) + pos];
        float offy = offs[((b * 18 + n + 9) << 12) + pos];
        float px = (float)(y + (n / 3)) + offx;   // (y+1) + (n/3 - 1) + off
        float py = (float)(x + (n % 3)) + offy;
        float fx = floorf(px), fy = floorf(py);
        float ltx = fminf(fmaxf(fx, 0.f), 63.f);
        float lty = fminf(fmaxf(fy, 0.f), 63.f);
        float rbx = fminf(fmaxf(fx + 1.f, 0.f), 63.f);
        float rby = fminf(fmaxf(fy + 1.f, 0.f), 63.f);
        float pxc = fminf(fmaxf(px, 0.f), 63.f);
        float pyc = fminf(fmaxf(py, 0.f), 63.f);
        float wlx = 1.f + ltx - pxc, wrx = 1.f - rbx + pxc;
        float wly = 1.f + lty - pyc, wry = 1.f - rby + pyc;
        sIdx[i * 4 + 0] = ((int)ltx << 6) + (int)lty;
        sIdx[i * 4 + 1] = ((int)rbx << 6) + (int)rby;
        sIdx[i * 4 + 2] = ((int)ltx << 6) + (int)rby;
        sIdx[i * 4 + 3] = ((int)rbx << 6) + (int)lty;
        sWt[i * 4 + 0] = wlx * wly;
        sWt[i * 4 + 1] = wrx * wry;
        sWt[i * 4 + 2] = wlx * wry;
        sWt[i * 4 + 3] = wrx * wly;
    }

    const int wid = tid >> 5, lane = tid & 31;
    const int wm = wid >> 2, wn = wid & 3;
    v8f acc[4];
    #pragma unroll
    for (int t = 0; t < 4; ++t) acc[t] = (v8f){0.f, 0.f, 0.f, 0.f, 0.f, 0.f, 0.f, 0.f};

    // per-lane bases: all fragment loads below use immediate offsets only
    const __bf16* bbase = pw + ((wn * 4 * 72) * 32 + lane) * 16;   // + nt*36864 + ktg*512
    const __bf16* abase = sA + (wm * 9 * 32 + lane) * 16;          // + kt*512

    for (int cb = 0; cb < 8; ++cb) {          // K chunks: 32 channels * 9 taps = 288
        __syncthreads();
        for (int e = tid; e < 9216; e += 256) {      // sample 32 rows x 288 K values
            int mi = e / 288, kk = e % 288;
            int c = (cb << 5) + kk / 9;
            int n = kk % 9;
            int b = (m0 + mi) >> 12;
            const float* hp = h + (((size_t)(b * 256 + c)) << 12);
            int mid = (mi * 9 + n) << 2;
            float val = sWt[mid]     * hp[sIdx[mid]]
                      + sWt[mid + 1] * hp[sIdx[mid + 1]]
                      + sWt[mid + 2] * hp[sIdx[mid + 2]]
                      + sWt[mid + 3] * hp[sIdx[mid + 3]];
            // write into A-fragment layout, contiguous k order (k = 16*half + elem) matching packed B
            int mt = mi >> 4, row = mi & 15;
            int kt = kk >> 5, kkk = kk & 31;
            int g = kkk >> 4, s = kkk & 15;
            int ln = (g << 4) + row;
            sA[((((mt * 9 + kt) << 5) + ln) << 4) + s] = (__bf16)val;
        }
        __syncthreads();

        // ---- software-pipelined WMMA sweep over 9 k-tiles ----
        const __bf16* bp = bbase + cb * 9 * 512;
        v16bf b0 = *(const v16bf*)(bp);
        v16bf b1 = *(const v16bf*)(bp + 36864);
        v16bf b2 = *(const v16bf*)(bp + 73728);
        v16bf b3 = *(const v16bf*)(bp + 110592);
        #pragma unroll
        for (int kt = 0; kt < 9; ++kt) {
            v16bf a = *(const v16bf*)(abase + kt * 512);
            v16bf c0, c1, c2, c3;
            if (kt < 8) {                       // prefetch next k-tile's B frags
                c0 = *(const v16bf*)(bp + (kt + 1) * 512);
                c1 = *(const v16bf*)(bp + 36864 + (kt + 1) * 512);
                c2 = *(const v16bf*)(bp + 73728 + (kt + 1) * 512);
                c3 = *(const v16bf*)(bp + 110592 + (kt + 1) * 512);
            }
            acc[0] = __builtin_amdgcn_wmma_f32_16x16x32_bf16(false, a, false, b0, (short)0, acc[0], false, false);
            acc[1] = __builtin_amdgcn_wmma_f32_16x16x32_bf16(false, a, false, b1, (short)0, acc[1], false, false);
            acc[2] = __builtin_amdgcn_wmma_f32_16x16x32_bf16(false, a, false, b2, (short)0, acc[2], false, false);
            acc[3] = __builtin_amdgcn_wmma_f32_16x16x32_bf16(false, a, false, b3, (short)0, acc[3], false, false);
            b0 = c0; b1 = c1; b2 = c2; b3 = c3;
        }
    }

    // C/D layout: VGPR r, lanes 0-15 -> M=r, lanes 16-31 -> M=r+8; N = lane&15
    // The 8 rows of one lane map to 8 consecutive pixels -> 2 x b128 stores per tile
    int mbase = m0 + (wm << 4) + ((lane >> 4) << 3);
    int b = mbase >> 12, posb = mbase & 4095;
    int nlo = lane & 15;
    #pragma unroll
    for (int nt = 0; nt < 4; ++nt) {
        int o = (wn << 6) + (nt << 4) + nlo;
        float* op = out + (((size_t)(b * 256 + o)) << 12) + posb;
        f4 lo = {acc[nt][0], acc[nt][1], acc[nt][2], acc[nt][3]};
        f4 hi = {acc[nt][4], acc[nt][5], acc[nt][6], acc[nt][7]};
        *(f4*)(op)     = lo;
        *(f4*)(op + 4) = hi;
    }
}

extern "C" void kernel_launch(void* const* d_in, const int* in_sizes, int n_in,
                              void* d_out, int out_size, void* d_ws, size_t ws_size,
                              hipStream_t stream) {
    const float* x   = (const float*)d_in[0];
    const float* w1  = (const float*)d_in[1];
    const float* b1  = (const float*)d_in[2];
    const float* w2  = (const float*)d_in[3];
    const float* b2  = (const float*)d_in[4];
    const float* pcw = (const float*)d_in[5];
    const float* pcb = (const float*)d_in[6];
    const float* dw  = (const float*)d_in[7];
    float* out = (float*)d_out;

    char* ws = (char*)d_ws;
    float*  h    = (float*)(ws);                 // 33,554,432 B : gated input
    float*  offs = (float*)(ws + 33554432);      //  2,359,296 B : offset maps
    __bf16* pw   = (__bf16*)(ws + 35913728);     //  1,179,648 B : packed B frags
    float*  avg  = (float*)(ws + 37093376);      //      8,192 B
    float*  mx   = (float*)(ws + 37101568);      //      8,192 B
    float*  att  = (float*)(ws + 37109760);      //      8,192 B

    BDCM_stats_kernel<<<2048, 256, 0, stream>>>(x, avg, mx);
    BDCM_gate_kernel<<<8, 256, 0, stream>>>(avg, mx, w1, b1, w2, b2, att);
    BDCM_scale_kernel<<<8192, 256, 0, stream>>>(x, att, h);
    BDCM_packw_kernel<<<2304, 256, 0, stream>>>(dw, pw);
    BDCM_offconv_kernel<<<512, 256, 0, stream>>>(h, pcw, pcb, offs);
    BDCM_deform_wmma_kernel<<<1024, 256, 0, stream>>>(h, offs, pw, out);
}
